// GraphConvolution_86466281603622
// MI455X (gfx1250) — matrix operand order repaired and checked by reading, not compiled, not run on hardware
//
#include <hip/hip_runtime.h>
#include <hip/hip_bf16.h>

typedef __attribute__((ext_vector_type(16))) __bf16 v16bf;
typedef __attribute__((ext_vector_type(8)))  __bf16 v8bf;
typedef __attribute__((ext_vector_type(4)))  __bf16 v4bf;
typedef __attribute__((ext_vector_type(8)))  float  v8f;

#define NODES 50000
#define BATCH 4
#define FEAT  256
#define MTOT  (BATCH * NODES)   // 200000 rows in the flattened GEMM
#define LDSK  264               // padded bf16 row stride (4-bank skew per row)

// ---------------- K1: weight -> transposed bf16 hi/lo ----------------
__global__ void prep_weight(const float* __restrict__ W,
                            __bf16* __restrict__ Wh, __bf16* __restrict__ Wl) {
    int k = blockIdx.x, n = threadIdx.x;         // W is [K=256][N=256]
    float x = W[k * FEAT + n];
    __bf16 h = (__bf16)x;
    __bf16 l = (__bf16)(x - (float)h);
    Wh[n * FEAT + k] = h;                        // transposed: [N][K]
    Wl[n * FEAT + k] = l;
}

// ---------------- K2: support = X @ W via bf16x3 WMMA ----------------
__global__ __launch_bounds__(256) void gemm_bf16x3(
    const float* __restrict__ X, const __bf16* __restrict__ Wh,
    const __bf16* __restrict__ Wl, float* __restrict__ S) {
    __shared__ __align__(16) __bf16 Ah[128 * LDSK];
    __shared__ __align__(16) __bf16 Al[128 * LDSK];

    const int tid = threadIdx.x;
    const int m0  = blockIdx.x * 128;
    const float4* X4 = (const float4*)X;

    // Stage 128x256 f32 A-tile into LDS as bf16 hi/lo (whole K at once).
    #pragma unroll
    for (int i = 0; i < 32; ++i) {
        int lin = tid + i * 256;                 // float4 index within tile
        int row = lin >> 6, c4 = lin & 63;
        int gr = m0 + row; if (gr >= MTOT) gr = MTOT - 1;
        float4 f = X4[(size_t)gr * 64 + c4];
        __bf16 h0 = (__bf16)f.x, h1 = (__bf16)f.y, h2 = (__bf16)f.z, h3 = (__bf16)f.w;
        v4bf h, l;
        h[0] = h0; h[1] = h1; h[2] = h2; h[3] = h3;
        l[0] = (__bf16)(f.x - (float)h0); l[1] = (__bf16)(f.y - (float)h1);
        l[2] = (__bf16)(f.z - (float)h2); l[3] = (__bf16)(f.w - (float)h3);
        *(v4bf*)&Ah[row * LDSK + c4 * 4] = h;
        *(v4bf*)&Al[row * LDSK + c4 * 4] = l;
    }
    __syncthreads();

    const int lane  = tid & 31;
    const int wm    = tid >> 5;                  // wave id = 16-row M strip
    const int lrow  = wm * 16 + (lane & 15);
    const int khalf = lane >> 4;                 // ISA A/B lane-half K select

    v8f zero = {0.f, 0.f, 0.f, 0.f, 0.f, 0.f, 0.f, 0.f};
    v8f acc[16];
    #pragma unroll
    for (int t = 0; t < 16; ++t) acc[t] = zero;

    for (int kk = 0; kk < 8; ++kk) {             // K in steps of 32
        const int k0 = kk * 32;
        const int ka = k0 + khalf * 8;           // 16-bit A layout: lane-half K octets
        v8bf a0h = *(const v8bf*)&Ah[lrow * LDSK + ka];
        v8bf a1h = *(const v8bf*)&Ah[lrow * LDSK + ka + 16];
        v8bf a0l = *(const v8bf*)&Al[lrow * LDSK + ka];
        v8bf a1l = *(const v8bf*)&Al[lrow * LDSK + ka + 16];
        v16bf ahi = __builtin_shufflevector(a0h, a1h, 0,1,2,3,4,5,6,7,8,9,10,11,12,13,14,15);
        v16bf alo = __builtin_shufflevector(a0l, a1l, 0,1,2,3,4,5,6,7,8,9,10,11,12,13,14,15);

        #pragma unroll
        for (int nt = 0; nt < 16; ++nt) {
            const int n = nt * 16 + (lane & 15); // B layout: same lane-half K octets as A
            const __bf16* bhp = &Wh[n * FEAT + k0 + khalf * 8];
            const __bf16* blp = &Wl[n * FEAT + k0 + khalf * 8];
            v8bf b0h = *(const v8bf*)bhp;
            v8bf b1h = *(const v8bf*)(bhp + 16);
            v8bf b0l = *(const v8bf*)blp;
            v8bf b1l = *(const v8bf*)(blp + 16);
            v16bf bhi = __builtin_shufflevector(b0h, b1h, 0,1,2,3,4,5,6,7,8,9,10,11,12,13,14,15);
            v16bf blo = __builtin_shufflevector(b0l, b1l, 0,1,2,3,4,5,6,7,8,9,10,11,12,13,14,15);

            acc[nt] = __builtin_amdgcn_wmma_f32_16x16x32_bf16(false, ahi, false, bhi,
                                                              (short)0, acc[nt], false, false);
            acc[nt] = __builtin_amdgcn_wmma_f32_16x16x32_bf16(false, ahi, false, blo,
                                                              (short)0, acc[nt], false, false);
            acc[nt] = __builtin_amdgcn_wmma_f32_16x16x32_bf16(false, alo, false, bhi,
                                                              (short)0, acc[nt], false, false);
        }
    }

    // C/D layout: VGPR j of lane l -> M = j + 8*(l>>4), N = l&15
    #pragma unroll
    for (int nt = 0; nt < 16; ++nt) {
        #pragma unroll
        for (int j = 0; j < 8; ++j) {
            int gr = m0 + wm * 16 + j + 8 * khalf;
            if (gr < MTOT) S[(size_t)gr * FEAT + nt * 16 + (lane & 15)] = acc[nt][j];
        }
    }
}

// ---------------- K3..K6: CSR build ----------------
__global__ void zero_counts(int* cnt, int n) {
    int i = blockIdx.x * 256 + threadIdx.x;
    if (i < n) cnt[i] = 0;
}

__global__ void count_edges(const int* __restrict__ rows, int* cnt, int E) {
    int e = blockIdx.x * 256 + threadIdx.x;
    if (e < E) atomicAdd(&cnt[rows[e]], 1);
}

__global__ __launch_bounds__(1024) void scan_rows(const int* __restrict__ cnt,
                                                  int* row_start, int* cursor, int n) {
    __shared__ int s[1024];
    int tid = threadIdx.x;
    int base = 0;
    for (int c0 = 0; c0 < n; c0 += 1024) {
        int idx = c0 + tid;
        int x = (idx < n) ? cnt[idx] : 0;
        s[tid] = x;
        __syncthreads();
        for (int off = 1; off < 1024; off <<= 1) {
            int v = (tid >= off) ? s[tid - off] : 0;
            __syncthreads();
            s[tid] += v;
            __syncthreads();
        }
        int incl = s[tid];
        if (idx < n) {
            int ex = base + incl - x;
            row_start[idx] = ex;
            cursor[idx]    = ex;
        }
        int tot = s[1023];
        __syncthreads();
        base += tot;
    }
    if (tid == 0) row_start[n] = base;
}

__global__ void fill_edges(const int* __restrict__ rows, const int* __restrict__ cols,
                           const float* __restrict__ vals, int* cursor,
                           int* __restrict__ ecol, float* __restrict__ evalv, int E) {
    int e = blockIdx.x * 256 + threadIdx.x;
    if (e < E) {
        int r = rows[e];
        int p = atomicAdd(&cursor[r], 1);
        ecol[p]  = cols[e];
        evalv[p] = vals[e];
    }
}

// ---------------- K7: gather aggregation + bias (no float atomics) ----------------
__global__ __launch_bounds__(256) void aggregate(
    const float* __restrict__ S, const int* __restrict__ row_start,
    const int* __restrict__ ecol, const float* __restrict__ evalv,
    const float* __restrict__ bias, float* __restrict__ out) {
    const int i   = blockIdx.x;                  // destination node
    const int tid = threadIdx.x;
    const int b   = tid >> 6;                    // batch plane
    const int f4  = tid & 63;                    // float4 column
    const float4* S4 = (const float4*)S;
    float4*       O4 = (float4*)out;
    const float4  bv = ((const float4*)bias)[f4];

    int e0 = row_start[i], e1 = row_start[i + 1];
    float4 acc = {0.f, 0.f, 0.f, 0.f};
    for (int e = e0; e < e1; ++e) {
        int   c = ecol[e];
        float v = evalv[e];
        if (e + 1 < e1)                           // global_prefetch next edge's row
            __builtin_prefetch(&S4[((size_t)b * NODES + ecol[e + 1]) * 64 + f4], 0, 0);
        float4 sv = S4[((size_t)b * NODES + c) * 64 + f4];
        acc.x += v * sv.x; acc.y += v * sv.y; acc.z += v * sv.z; acc.w += v * sv.w;
    }
    O4[((size_t)b * NODES + i) * 64 + f4] =
        make_float4(acc.x + bv.x, acc.y + bv.y, acc.z + bv.z, acc.w + bv.w);
}

// ---------------- host launcher ----------------
extern "C" void kernel_launch(void* const* d_in, const int* in_sizes, int n_in,
                              void* d_out, int out_size, void* d_ws, size_t ws_size,
                              hipStream_t stream) {
    const float* X    = (const float*)d_in[0];
    const int*   rows = (const int*)d_in[1];
    const int*   cols = (const int*)d_in[2];
    const float* vals = (const float*)d_in[3];
    const float* W    = (const float*)d_in[4];
    const float* bias = (const float*)d_in[5];
    const int E = in_sizes[1];

    char* ws = (char*)d_ws;
    size_t off = 0;
    auto alloc = [&](size_t bytes) -> void* {
        void* p = ws + off;
        off += (bytes + 255) & ~(size_t)255;
        return p;
    };
    float*  S         = (float*)alloc((size_t)MTOT * FEAT * 4);   // 204.8 MB
    __bf16* Wh        = (__bf16*)alloc((size_t)FEAT * FEAT * 2);
    __bf16* Wl        = (__bf16*)alloc((size_t)FEAT * FEAT * 2);
    int*    cnt       = (int*)alloc((size_t)NODES * 4);
    int*    row_start = (int*)alloc((size_t)(NODES + 1) * 4);
    int*    cursor    = (int*)alloc((size_t)NODES * 4);
    int*    ecol      = (int*)alloc((size_t)E * 4);
    float*  evalv     = (float*)alloc((size_t)E * 4);

    prep_weight<<<FEAT, FEAT, 0, stream>>>(W, Wh, Wl);
    gemm_bf16x3<<<(MTOT + 127) / 128, 256, 0, stream>>>(X, Wh, Wl, S);
    zero_counts<<<(NODES + 255) / 256, 256, 0, stream>>>(cnt, NODES);
    count_edges<<<(E + 255) / 256, 256, 0, stream>>>(rows, cnt, E);
    scan_rows<<<1, 1024, 0, stream>>>(cnt, row_start, cursor, NODES);
    fill_edges<<<(E + 255) / 256, 256, 0, stream>>>(rows, cols, vals, cursor, ecol, evalv, E);
    aggregate<<<NODES, 256, 0, stream>>>(S, row_start, ecol, evalv, bias, (float*)d_out);
}
